// BasisResidualFFN_7361573945459
// MI455X (gfx1250) — compile-verified, matrix operand order checked by reading, not run on hardware
//
#include <hip/hip_runtime.h>
#include <math.h>

typedef __bf16 bf16_t;
typedef bf16_t v16bf __attribute__((ext_vector_type(16)));
typedef float  v8f   __attribute__((ext_vector_type(8)));
typedef int    brf_v4i __attribute__((vector_size(16)));

#if defined(__gfx1250__) && \
    __has_builtin(__builtin_amdgcn_global_load_async_to_lds_b128) && \
    __has_builtin(__builtin_amdgcn_s_wait_asynccnt)
#define BRF_ASYNC_LDS 1
#else
#define BRF_ASYNC_LDS 0
#endif

namespace {
constexpr int kB   = 2;
constexpr int kS   = 2048;
constexpr int kT   = kB * kS;      // 4096 tokens
constexpr int kD   = 1024;
constexpr int kK   = 8;            // selected neurons per token
constexpr int kNB  = 16;           // n_basis
constexpr int kR   = 32;           // rank
constexpr int kDFF = 4096;
constexpr int kJ   = kNB * kR;     // 512 = flattened (n, r)
constexpr int kLdsStride = 40;     // 32 K-halves + 8 pad (bank-conflict-free)
}

// ---------------------------------------------------------------------------
// Step 1: token_recipe[t][n] = sum_k weight[t][k] * softmax(recipe[idx[t][k]])[n]
// ---------------------------------------------------------------------------
__global__ void brf_recipe_kernel(const int* __restrict__ nidx,
                                  const float* __restrict__ nw,
                                  const float* __restrict__ recipe,
                                  float* __restrict__ tr) {
  int t = blockIdx.x * blockDim.x + threadIdx.x;
  if (t >= kT) return;
  float acc[kNB];
#pragma unroll
  for (int n = 0; n < kNB; ++n) acc[n] = 0.f;
  for (int k = 0; k < kK; ++k) {
    int id = nidx[t * kK + k];
    float wk = nw[t * kK + k];
    const float* row = recipe + id * kNB;
    float v[kNB];
    float m = -3.0e38f;
#pragma unroll
    for (int n = 0; n < kNB; ++n) { v[n] = row[n]; m = fmaxf(m, v[n]); }
    float s = 0.f;
#pragma unroll
    for (int n = 0; n < kNB; ++n) { v[n] = expf(v[n] - m); s += v[n]; }
    float scale = wk / s;
#pragma unroll
    for (int n = 0; n < kNB; ++n) acc[n] += v[n] * scale;
  }
#pragma unroll
  for (int n = 0; n < kNB; ++n) tr[t * kNB + n] = acc[n];
}

// ---------------------------------------------------------------------------
// Elementwise fp32 -> bf16
// ---------------------------------------------------------------------------
__global__ void brf_f2bf_kernel(const float* __restrict__ in,
                                bf16_t* __restrict__ out, int n) {
  int i = blockIdx.x * blockDim.x + threadIdx.x;
  if (i < n) out[i] = (bf16_t)in[i];
}

// ---------------------------------------------------------------------------
// Transpose fp32 [rows=K][cols=N] -> bf16 N-major [N][K]
// ---------------------------------------------------------------------------
__global__ void brf_wT_kernel(const float* __restrict__ in,
                              bf16_t* __restrict__ out, int rows, int cols) {
  int i = blockIdx.x * blockDim.x + threadIdx.x;
  if (i >= rows * cols) return;
  int n = i / rows;
  int k = i - n * rows;
  out[i] = (bf16_t)in[k * cols + n];
}

// ---------------------------------------------------------------------------
// basis_A [n][d][r] fp32 -> two bf16 B-operand layouts:
//  Ab1[j][d]  (B for GEMM1: K=D, N=J, j = n*32+r)
//  Ab2[d][j]  (B for GEMM2: K=J, N=D)
// ---------------------------------------------------------------------------
__global__ void brf_basis_kernel(const float* __restrict__ bA,
                                 bf16_t* __restrict__ Ab1,
                                 bf16_t* __restrict__ Ab2) {
  int i = blockIdx.x * blockDim.x + threadIdx.x;
  if (i >= kNB * kD * kR) return;
  int r = i % kR;
  int d = (i / kR) % kD;
  int n = i / (kR * kD);
  float v = bA[i];
  int j = n * kR + r;
  Ab1[(size_t)j * kD + d] = (bf16_t)v;
  Ab2[(size_t)d * kJ + j] = (bf16_t)v;
}

// ---------------------------------------------------------------------------
//   h[t][r] = sum_n tr[t][n] * P[t][n*32+r]
//   g[t][n*32+r] = bf16( tr[t][n] * h[t][r] )
// ---------------------------------------------------------------------------
__global__ void brf_mid_kernel(const float* __restrict__ P,
                               const float* __restrict__ tr,
                               bf16_t* __restrict__ g) {
  int gid = blockIdx.x * blockDim.x + threadIdx.x;
  int t = gid >> 5;
  int r = gid & 31;
  if (t >= kT) return;
  const float* pt = P + (size_t)t * kJ;
  const float* trt = tr + (size_t)t * kNB;
  float h = 0.f;
#pragma unroll
  for (int n = 0; n < kNB; ++n) h += trt[n] * pt[n * kR + r];
#pragma unroll
  for (int n = 0; n < kNB; ++n)
    g[(size_t)t * kJ + n * kR + r] = (bf16_t)(trt[n] * h);
}

// ---------------------------------------------------------------------------
// WMMA bf16 GEMM:  C[M,N] = A[M,K] * B[K,N], B stored N-major (Bt[n][k]).
// Block: 128 threads = 4 waves. Block tile 128(M) x 64(N).
// Per wave: 32 rows x 64 cols = 8 v_wmma_f32_16x16x32_bf16 per K-step.
// B tile (64 cols x 32 K, shared by all 4 waves) staged in LDS, double-
// buffered via gfx1250 async global->LDS (ASYNCcnt). A fragments are
// software-pipelined in registers (load K-step i+1 while computing i).
// ---------------------------------------------------------------------------
enum { EPI_P = 0, EPI_XF = 1, EPI_GELU = 2, EPI_OUT = 3 };

union BrfFrag { v16bf v; uint4 u[2]; };

__global__ __launch_bounds__(128)
void brf_gemm_wmma_bf16(const bf16_t* __restrict__ A,
                        const bf16_t* __restrict__ Bt,
                        int M, int N, int K, int epi,
                        float* __restrict__ outF,
                        bf16_t* __restrict__ outH,
                        const float* __restrict__ bias,
                        const float* __restrict__ xres,
                        const float* __restrict__ alphaPtr) {
  __shared__ bf16_t ldsB[2][64 * kLdsStride];   // 2 x 2560 bf16 = 10 KB

  const int tid  = threadIdx.x;
  const int lane = tid & 31;
  const int wave = tid >> 5;
  const int half = lane >> 4;   // 0: K chunks {0..7,16..23}, 1: {8..15,24..31}
  const int r16  = lane & 15;

  const int m0 = blockIdx.y * 128 + wave * 32;
  const int n0 = blockIdx.x * 64;

  // Cooperative B-tile fill: 64 rows x 4 x 16B chunks = 256 chunks,
  // 2 chunks per thread. Each async instruction moves 16B/lane.
  const int chunkRow0   = (tid * 2) >> 2;
  const int chunkPiece0 = (tid * 2) & 3;
  const bf16_t* bBase = Bt + (size_t)(n0 + chunkRow0) * K + chunkPiece0 * 8;
  bf16_t* lBase0 = &ldsB[0][chunkRow0 * kLdsStride + chunkPiece0 * 8];
  bf16_t* lBase1 = &ldsB[1][chunkRow0 * kLdsStride + chunkPiece0 * 8];

  auto issue_tile = [&](int buf, int kk) {
    bf16_t* lb = buf ? lBase1 : lBase0;
#pragma unroll
    for (int c = 0; c < 2; ++c) {
      const bf16_t* src = bBase + kk + c * 8;
      bf16_t* dst = lb + c * 8;
#if BRF_ASYNC_LDS
      __builtin_amdgcn_global_load_async_to_lds_b128(
          (__attribute__((address_space(1))) brf_v4i*)src,
          (__attribute__((address_space(3))) brf_v4i*)dst, 0, 0);
#else
      *(uint4*)dst = *(const uint4*)src;
#endif
    }
  };

  v8f acc[2][4];
#pragma unroll
  for (int h = 0; h < 2; ++h)
#pragma unroll
    for (int t = 0; t < 4; ++t)
      acc[h][t] = (v8f){0.f, 0.f, 0.f, 0.f, 0.f, 0.f, 0.f, 0.f};

  const bf16_t* aRow0 = A + (size_t)(m0 + r16) * K;
  const bf16_t* aRow1 = A + (size_t)(m0 + 16 + r16) * K;

  auto load_afrag = [&](BrfFrag af[2], int kk) {
    af[0].u[0] = *reinterpret_cast<const uint4*>(aRow0 + kk + half * 8);
    af[0].u[1] = *reinterpret_cast<const uint4*>(aRow0 + kk + 16 + half * 8);
    af[1].u[0] = *reinterpret_cast<const uint4*>(aRow1 + kk + half * 8);
    af[1].u[1] = *reinterpret_cast<const uint4*>(aRow1 + kk + 16 + half * 8);
  };

  issue_tile(0, 0);

  BrfFrag afCur[2], afNext[2];
  load_afrag(afCur, 0);

  for (int kk = 0; kk < K; kk += 32) {
    const int cur = (kk >> 5) & 1;
    const bool more = (kk + 32 < K);
    if (more) {
      issue_tile(cur ^ 1, kk + 32);
#if BRF_ASYNC_LDS
      __builtin_amdgcn_s_wait_asynccnt(2);   // current B tile's 2 ops retired
#endif
    } else {
#if BRF_ASYNC_LDS
      __builtin_amdgcn_s_wait_asynccnt(0);
#endif
    }
    __syncthreads();

    // A software pipeline: start next K-step's global loads now; the
    // WMMAs below depend only on LDS (dscnt), not on these.
    if (more) load_afrag(afNext, kk + 32);

#pragma unroll
    for (int t = 0; t < 4; ++t) {
      const int col = t * 16 + r16;
      const bf16_t* bp = &ldsB[cur][col * kLdsStride];
      BrfFrag bf;
      bf.u[0] = *reinterpret_cast<const uint4*>(bp + half * 8);        // ds_load_b128
      bf.u[1] = *reinterpret_cast<const uint4*>(bp + 16 + half * 8);
      acc[0][t] = __builtin_amdgcn_wmma_f32_16x16x32_bf16(
          false, afCur[0].v, false, bf.v, (short)0, acc[0][t], false, false);
      acc[1][t] = __builtin_amdgcn_wmma_f32_16x16x32_bf16(
          false, afCur[1].v, false, bf.v, (short)0, acc[1][t], false, false);
    }
    __syncthreads();   // all reads of ldsB[cur] done before it is refilled

    if (more) { afCur[0] = afNext[0]; afCur[1] = afNext[1]; }
  }

  const float alpha = alphaPtr[0];

#pragma unroll
  for (int h = 0; h < 2; ++h) {
#pragma unroll
    for (int t = 0; t < 4; ++t) {
#pragma unroll
      for (int i = 0; i < 8; ++i) {
        // C/D layout: VGPR i -> row (half*8 + i), lane r16 -> col
        int row = m0 + h * 16 + half * 8 + i;
        int col = n0 + t * 16 + r16;
        size_t o = (size_t)row * N + col;
        float v = acc[h][t][i];
        if (epi == EPI_P) {
          outF[o] = v;
        } else if (epi == EPI_XF) {
          outH[o] = (bf16_t)(xres[o] + alpha * v);
        } else if (epi == EPI_GELU) {
          float z = v + bias[col];
          outH[o] = (bf16_t)(0.5f * z * (1.0f + erff(z * 0.70710678118654752f)));
        } else {  // EPI_OUT
          outF[o] = v + bias[col];
        }
      }
    }
  }
}

// ---------------------------------------------------------------------------
// Host launcher
// ---------------------------------------------------------------------------
extern "C" void kernel_launch(void* const* d_in, const int* in_sizes, int n_in,
                              void* d_out, int out_size, void* d_ws, size_t ws_size,
                              hipStream_t stream) {
  const float* x      = (const float*)d_in[0];
  const int*   nidx   = (const int*)d_in[1];
  const float* nw     = (const float*)d_in[2];
  const float* recipe = (const float*)d_in[3];
  const float* basisA = (const float*)d_in[4];
  const float* wup    = (const float*)d_in[5];
  const float* wupb   = (const float*)d_in[6];
  const float* wdn    = (const float*)d_in[7];
  const float* wdnb   = (const float*)d_in[8];
  const float* alpha  = (const float*)d_in[9];
  float* out = (float*)d_out;

  char* ws = (char*)d_ws;
  size_t off = 0;
  auto alloc = [&](size_t bytes) -> void* {
    void* p = ws + off;
    off += (bytes + 255) & ~(size_t)255;
    return p;
  };
  float*  tr   = (float*)alloc((size_t)kT * kNB * 4);
  bf16_t* xb   = (bf16_t*)alloc((size_t)kT * kD * 2);
  bf16_t* Ab1  = (bf16_t*)alloc((size_t)kJ * kD * 2);
  bf16_t* Ab2  = (bf16_t*)alloc((size_t)kD * kJ * 2);
  float*  P    = (float*)alloc((size_t)kT * kJ * 4);
  bf16_t* g    = (bf16_t*)alloc((size_t)kT * kJ * 2);
  bf16_t* xf   = (bf16_t*)alloc((size_t)kT * kD * 2);
  bf16_t* wupT = (bf16_t*)alloc((size_t)kDFF * kD * 2);
  bf16_t* hid  = (bf16_t*)alloc((size_t)kT * kDFF * 2);
  bf16_t* wdnT = (bf16_t*)alloc((size_t)kD * kDFF * 2);
  (void)ws_size; (void)in_sizes; (void)n_in; (void)out_size;

  // 1) token recipes
  brf_recipe_kernel<<<(kT + 127) / 128, 128, 0, stream>>>(nidx, nw, recipe, tr);

  // 2) conversions
  {
    int n = kT * kD;
    brf_f2bf_kernel<<<(n + 255) / 256, 256, 0, stream>>>(x, xb, n);
  }
  {
    int n = kNB * kD * kR;
    brf_basis_kernel<<<(n + 255) / 256, 256, 0, stream>>>(basisA, Ab1, Ab2);
  }
  {
    int n = kD * kDFF;
    brf_wT_kernel<<<(n + 255) / 256, 256, 0, stream>>>(wup, wupT, kD, kDFF);
    brf_wT_kernel<<<(n + 255) / 256, 256, 0, stream>>>(wdn, wdnT, kDFF, kD);
  }

  dim3 blk(128);
  // 3) P = xb @ A_r          [T x J], K = D
  brf_gemm_wmma_bf16<<<dim3(kJ / 64, kT / 128), blk, 0, stream>>>(
      xb, Ab1, kT, kJ, kD, EPI_P, P, nullptr, nullptr, nullptr, alpha);

  // 4) h/g
  brf_mid_kernel<<<(kT * 32) / 128, 128, 0, stream>>>(P, tr, g);

  // 5) xf = x + alpha * (g @ A_r^T)   [T x D], K = J
  brf_gemm_wmma_bf16<<<dim3(kD / 64, kT / 128), blk, 0, stream>>>(
      g, Ab2, kT, kD, kJ, EPI_XF, nullptr, xf, nullptr, x, alpha);

  // 6) hid = gelu(xf @ w_up + b_up)   [T x DFF], K = D
  brf_gemm_wmma_bf16<<<dim3(kDFF / 64, kT / 128), blk, 0, stream>>>(
      xf, wupT, kT, kDFF, kD, EPI_GELU, nullptr, hid, wupb, nullptr, alpha);

  // 7) out = hid @ w_down + b_down    [T x D], K = DFF
  brf_gemm_wmma_bf16<<<dim3(kD / 64, kT / 128), blk, 0, stream>>>(
      hid, wdnT, kT, kD, kDFF, EPI_OUT, out, nullptr, wdnb, nullptr, alpha);
}
